// Model_17093969838448
// MI455X (gfx1250) — matrix operand order repaired and checked by reading, not compiled, not run on hardware
//
#include <hip/hip_runtime.h>
#include <cstdint>

typedef __attribute__((ext_vector_type(16))) __bf16 v16bf;
typedef __attribute__((ext_vector_type(8)))  __bf16 v8bf;
typedef __attribute__((ext_vector_type(8)))  float  v8f;
typedef __attribute__((ext_vector_type(4)))  unsigned int v4ui;
typedef __attribute__((ext_vector_type(8)))  int v8i;
typedef __attribute__((ext_vector_type(4)))  int v4i;

union FragAB { v16bf v; v8bf h[2]; };

#define D_DIM 1024
#define H_DIM 512
#define L_DIM 60

__device__ __forceinline__ unsigned short f2bf(float f) {
    unsigned int u = __float_as_uint(f);
    unsigned int r = u + 0x7FFFu + ((u >> 16) & 1u);   // round-to-nearest-even
    return (unsigned short)(r >> 16);
}

// ---- Tensor Data Mover: 2-D bf16 tile load Global->LDS with row padding ----
// Tile: tile_d1 lines of tile_d0 halfs, source pitch = tensor_d0 halfs.
// LDS rows padded to tile_d0 + 8 halfs (pad_interval=16 DWORDs, pad_amount=4 DWORDs).
__device__ __forceinline__ void tdm_load_tile_bf16(
    unsigned lds_off, const unsigned short* gsrc,
    unsigned tensor_d0, unsigned tensor_d1,
    unsigned tile_d0, unsigned tile_d1)
{
    const unsigned long long ga = (unsigned long long)(size_t)gsrc;
    v4ui g0;
    g0[0] = 1u;                                            // count=1 (valid user D#)
    g0[1] = lds_off;                                       // lds_addr (bytes)
    g0[2] = (unsigned)ga;                                  // global_addr[31:0]
    g0[3] = (unsigned)((ga >> 32) & 0x01FFFFFFu) | (2u << 30); // addr[56:32] | type=2
    v8i g1;
    g1[0] = (int)((1u << 16)      // data_size = 2 bytes
                | (1u << 20)      // pad_enable
                | (3u << 22)      // pad_interval: 16 DWORDs
                | (3u << 25));    // pad_amount:   4 DWORDs
    g1[1] = (int)((tensor_d0 & 0xFFFFu) << 16);            // tensor_dim0[15:0]
    g1[2] = (int)((tensor_d0 >> 16) | ((tensor_d1 & 0xFFFFu) << 16));
    g1[3] = (int)((tensor_d1 >> 16) | (tile_d0 << 16));    // tile_dim0
    g1[4] = (int)(tile_d1 & 0xFFFFu);                      // tile_dim1 (tile_dim2=0)
    g1[5] = (int)tensor_d0;                                // tensor_dim0_stride[31:0]
    g1[6] = 0;                                             // stride hi / dim1_stride lo
    g1[7] = 0;
    const v4i zero4 = {0, 0, 0, 0};
    const v8i zero8 = {0, 0, 0, 0, 0, 0, 0, 0};
    __builtin_amdgcn_tensor_load_to_lds(g0, g1, zero4, zero4, zero8, 0);
}

// ---------- small utility kernels ----------
__global__ void k_zero(float* p, int n) {
    int i = blockIdx.x * blockDim.x + threadIdx.x;
    if (i < n) p[i] = 0.f;
}

__global__ void k_deg(const int* __restrict__ col, float* __restrict__ deg, int e) {
    int i = blockIdx.x * blockDim.x + threadIdx.x;
    if (i < e) atomicAdd(&deg[col[i]], 1.0f);
}

__global__ void k_dis(const float* __restrict__ deg, float* __restrict__ dis, int n) {
    int i = blockIdx.x * blockDim.x + threadIdx.x;
    if (i < n) dis[i] = rsqrtf(fmaxf(deg[i], 1.0f));
}

__global__ void k_f2bf(const float* __restrict__ src, unsigned short* __restrict__ dst, int n) {
    int i = blockIdx.x * blockDim.x + threadIdx.x;
    if (i < n) dst[i] = f2bf(src[i]);
}

// ---------- one-time W1 transpose: W1T[n][k] = bf16(W1[k][n]) ----------
__global__ __launch_bounds__(256) void k_transposeW(const float* __restrict__ W1,
                                                    unsigned short* __restrict__ W1T)
{
    __shared__ float tile[32][33];
    const int n0 = (blockIdx.x & 15) * 32;     // H/32 = 16
    const int k0 = (blockIdx.x >> 4) * 32;     // D/32 = 32
    const int tx = threadIdx.x & 31, ty = threadIdx.x >> 5;
    #pragma unroll
    for (int j = 0; j < 32; j += 8)
        tile[ty + j][tx] = W1[(size_t)(k0 + ty + j) * H_DIM + n0 + tx];
    __syncthreads();
    #pragma unroll
    for (int j = 0; j < 32; j += 8)
        W1T[(size_t)(n0 + ty + j) * D_DIM + k0 + tx] = f2bf(tile[tx][ty + j]);
}

// ---------- main GEMM: x1 = feat @ W1 + b1 (bf16 WMMA, TDM-staged tiles) ----------
// 256 threads (8 waves), 64x64 tile, K staged 32/step, TDM double-buffered.
__global__ __launch_bounds__(256) void k_gemm_x1(
    const unsigned short* __restrict__ Abf,   // N x 1024, bf16
    const unsigned short* __restrict__ W1T,   // 512 x 1024, bf16 (n-major)
    const float* __restrict__ b1,
    float* __restrict__ x1, float* __restrict__ gnn,
    int nBlocksM, int Ntot)
{
    __shared__ __align__(16) unsigned short As[2][64 * 40];  // [m][k], stride 40 halfs
    __shared__ __align__(16) unsigned short Bt[2][64 * 40];  // [n][k], stride 40 halfs

    const int bid = blockIdx.x;
    const int bm = bid % nBlocksM, bn = bid / nBlocksM;
    const int rowBase = bm * 64, colBase = bn * 64;
    const int t = threadIdx.x;
    const int w = t >> 5, lane = t & 31, g = lane >> 4, l16 = lane & 15;
    const int mt = w & 3, ntBase = (w >> 2) * 2;

    const unsigned asOff = (unsigned)(size_t)(void*)&As[0][0];
    const unsigned btOff = (unsigned)(size_t)(void*)&Bt[0][0];

    v8f acc0, acc1;
    #pragma unroll
    for (int i = 0; i < 8; ++i) { acc0[i] = 0.f; acc1[i] = 0.f; }

    const int NSTEPS = D_DIM / 32;

    if (t < 32) {   // wave 0 drives the TDM
        tdm_load_tile_bf16(asOff, Abf + (size_t)rowBase * D_DIM, D_DIM, (unsigned)Ntot, 32, 64);
        tdm_load_tile_bf16(btOff, W1T + (size_t)colBase * D_DIM, D_DIM, H_DIM, 32, 64);
    }

    for (int i = 0; i < NSTEPS; ++i) {
        const int cur = i & 1;
        if (t < 32) {
            if (i + 1 < NSTEPS) {
                const unsigned buf = (unsigned)(cur ^ 1) * 64u * 40u * 2u;
                const unsigned k0 = (unsigned)(i + 1) * 32u;
                tdm_load_tile_bf16(asOff + buf, Abf + (size_t)rowBase * D_DIM + k0,
                                   D_DIM, (unsigned)Ntot, 32, 64);
                tdm_load_tile_bf16(btOff + buf, W1T + (size_t)colBase * D_DIM + k0,
                                   D_DIM, H_DIM, 32, 64);
                __builtin_amdgcn_s_wait_tensorcnt(2);   // current tile's 2 loads done
            } else {
                __builtin_amdgcn_s_wait_tensorcnt(0);
            }
        }
        __syncthreads();

        // A fragment: elems 0-7 -> K=8g..8g+7, elems 8-15 -> K=8g+16..8g+23
        FragAB a, b0, b1f;
        const int m = mt * 16 + l16;
        a.h[0] = *(const v8bf*)&As[cur][m * 40 + 8 * g];
        a.h[1] = *(const v8bf*)&As[cur][m * 40 + 8 * g + 16];
        // B fragment (transposed tile): elems 0-7 -> K=8g..8g+7, elems 8-15 -> K=8g+16..8g+23
        const int n0r = (ntBase + 0) * 16 + l16;
        const int n1r = (ntBase + 1) * 16 + l16;
        b0.h[0]  = *(const v8bf*)&Bt[cur][n0r * 40 + 8 * g];
        b0.h[1]  = *(const v8bf*)&Bt[cur][n0r * 40 + 8 * g + 16];
        b1f.h[0] = *(const v8bf*)&Bt[cur][n1r * 40 + 8 * g];
        b1f.h[1] = *(const v8bf*)&Bt[cur][n1r * 40 + 8 * g + 16];

        acc0 = __builtin_amdgcn_wmma_f32_16x16x32_bf16(false, a.v, false, b0.v,  (short)0, acc0, false, false);
        acc1 = __builtin_amdgcn_wmma_f32_16x16x32_bf16(false, a.v, false, b1f.v, (short)0, acc1, false, false);
        __syncthreads();
    }

    // epilogue: C/D layout -> lane l, vgpr i: M = i + 8*(l>>4), N = l&15
    #pragma unroll
    for (int j = 0; j < 2; ++j) {
        v8f acc = j ? acc1 : acc0;
        const int n = colBase + (ntBase + j) * 16 + l16;
        const float bias = b1[n];
        #pragma unroll
        for (int i = 0; i < 8; ++i) {
            const int m = rowBase + mt * 16 + i + 8 * g;
            const float v = acc[i] + bias;
            x1[(size_t)m * H_DIM + n]  = v;
            gnn[(size_t)m * H_DIM + n] = v;
        }
    }
}

// ---------- per-node projections: s = gnn@Wi, t = gnn@Wj ----------
__global__ __launch_bounds__(256) void k_st(
    const float* __restrict__ gnn, const float* __restrict__ Wg,
    float* __restrict__ s, float* __restrict__ t_, int layer, int n)
{
    const int wave = threadIdx.x >> 5, lane = threadIdx.x & 31;
    const int node = blockIdx.x * 8 + wave;
    if (node >= n) return;
    const float* Wi = Wg + layer * (2 * H_DIM);
    const float* Wj = Wi + H_DIM;
    const float* gr = gnn + (size_t)node * H_DIM;
    float sv = 0.f, tv = 0.f;
    #pragma unroll
    for (int j = 0; j < 16; ++j) {
        const int idx = lane + 32 * j;
        const float gv = gr[idx];
        sv += gv * Wi[idx];
        tv += gv * Wj[idx];
    }
    #pragma unroll
    for (int off = 16; off > 0; off >>= 1) {
        sv += __shfl_xor(sv, off, 32);
        tv += __shfl_xor(tv, off, 32);
    }
    if (lane == 0) { s[node] = sv; t_[node] = tv; }
}

// ---------- one GNN layer: per-dialog agg = A(64x64) @ gnn_block(64x256) ----------
__global__ __launch_bounds__(256) void k_layer(
    float* __restrict__ gnn, const float* __restrict__ s, const float* __restrict__ t_,
    const float* __restrict__ dis, const float* __restrict__ bg, int layer)
{
    __shared__ __align__(16) unsigned short Abuf[64 * 72];   // [c][r] bf16
    __shared__ __align__(16) unsigned short Gst[256 * 72];   // [n][r] bf16

    const int gb = blockIdx.x >> 1;
    const int n0 = (blockIdx.x & 1) * 256;
    const int t = threadIdx.x;
    const int w = t >> 5, lane = t & 31, g = lane >> 4, l16 = lane & 15;
    const float bgv = bg[layer];
    const int base = gb * L_DIM;

    #pragma unroll
    for (int it = 0; it < 16; ++it) {
        const int id = t + 256 * it;
        const int c = id >> 6, r = id & 63;
        float v = 0.f;
        if (c < L_DIM && r < L_DIM && r != c) {
            const float nrm = dis[base + r] * dis[base + c];
            v = nrm * tanhf(s[base + c] + t_[base + r] + bgv);
        }
        Abuf[c * 72 + r] = f2bf(v);
    }
    for (int it = 0; it < 64; ++it) {
        const int id = t + 256 * it;
        const int r = id >> 8, n = id & 255;
        const float v = (r < L_DIM) ? gnn[(size_t)(base + r) * H_DIM + n0 + n] : 0.f;
        Gst[n * 72 + r] = f2bf(v);
    }
    __syncthreads();

    const int mt = w & 3;
    #pragma unroll
    for (int j = 0; j < 8; ++j) {
        const int nt = (w >> 2) * 8 + j;
        v8f acc;
        #pragma unroll
        for (int i = 0; i < 8; ++i) acc[i] = 0.f;
        #pragma unroll
        for (int ks = 0; ks < 2; ++ks) {
            FragAB a, b;
            const int m = mt * 16 + l16;
            a.h[0] = *(const v8bf*)&Abuf[m * 72 + ks * 32 + 8 * g];
            a.h[1] = *(const v8bf*)&Abuf[m * 72 + ks * 32 + 8 * g + 16];
            const int nn = nt * 16 + l16;
            b.h[0] = *(const v8bf*)&Gst[nn * 72 + ks * 32 + 8 * g];
            b.h[1] = *(const v8bf*)&Gst[nn * 72 + ks * 32 + 8 * g + 16];
            acc = __builtin_amdgcn_wmma_f32_16x16x32_bf16(false, a.v, false, b.v, (short)0, acc, false, false);
        }
        #pragma unroll
        for (int i = 0; i < 8; ++i) {
            const int M = mt * 16 + i + 8 * g;
            if (M < L_DIM) {
                const size_t idx = (size_t)(base + M) * H_DIM + n0 + nt * 16 + l16;
                gnn[idx] += acc[i];   // disjoint coords: in-place safe
            }
        }
    }
}

// ---------- out = concat([x1, gnn], axis=1) ----------
__global__ void k_concat(const float* __restrict__ x1, const float* __restrict__ gnn,
                         float* __restrict__ out, int n) {
    int i = blockIdx.x * blockDim.x + threadIdx.x;
    if (i < n * 1024) {
        const int m = i >> 10, h = i & 1023;
        out[i] = (h < H_DIM) ? x1[(size_t)m * H_DIM + h]
                             : gnn[(size_t)m * H_DIM + (h - H_DIM)];
    }
}

extern "C" void kernel_launch(void* const* d_in, const int* in_sizes, int n_in,
                              void* d_out, int out_size, void* d_ws, size_t ws_size,
                              hipStream_t stream)
{
    const float* feat = (const float*)d_in[0];
    const float* W1   = (const float*)d_in[1];
    const float* b1   = (const float*)d_in[2];
    const float* Wg   = (const float*)d_in[3];
    const float* bg   = (const float*)d_in[4];
    const int*   row  = (const int*)d_in[5];
    const int*   col  = (const int*)d_in[6];
    (void)row; (void)n_in; (void)out_size; (void)ws_size;

    const int N = in_sizes[0] / D_DIM;   // 3840
    const int E = in_sizes[5];           // 226560

    char* ws = (char*)d_ws;
    size_t off = 0;
    auto alloc = [&](size_t bytes) -> char* {
        char* p = ws + off;
        off = (off + bytes + 255) & ~(size_t)255;
        return p;
    };
    unsigned short* Abf = (unsigned short*)alloc((size_t)N * D_DIM * 2);
    unsigned short* W1T = (unsigned short*)alloc((size_t)H_DIM * D_DIM * 2);
    float* x1  = (float*)alloc((size_t)N * H_DIM * 4);
    float* gnn = (float*)alloc((size_t)N * H_DIM * 4);
    float* s   = (float*)alloc((size_t)N * 4);
    float* t_  = (float*)alloc((size_t)N * 4);
    float* deg = (float*)alloc((size_t)N * 4);
    float* dis = (float*)alloc((size_t)N * 4);

    k_zero<<<(N + 255) / 256, 256, 0, stream>>>(deg, N);
    k_deg<<<(E + 255) / 256, 256, 0, stream>>>(col, deg, E);
    k_dis<<<(N + 255) / 256, 256, 0, stream>>>(deg, dis, N);
    k_f2bf<<<(N * D_DIM + 255) / 256, 256, 0, stream>>>(feat, Abf, N * D_DIM);
    k_transposeW<<<(H_DIM / 32) * (D_DIM / 32), 256, 0, stream>>>(W1, W1T);

    const int nBlocksM = N / 64;
    k_gemm_x1<<<nBlocksM * (H_DIM / 64), 256, 0, stream>>>(Abf, W1T, b1, x1, gnn, nBlocksM, N);

    const int nd = N / L_DIM;
    for (int k = 0; k < 4; ++k) {
        k_st<<<(N + 7) / 8, 256, 0, stream>>>(gnn, Wg, s, t_, k, N);
        k_layer<<<nd * 2, 256, 0, stream>>>(gnn, s, t_, dis, bg, k);
    }
    k_concat<<<(N * 1024 + 255) / 256, 256, 0, stream>>>(x1, gnn, (float*)d_out, N);
}
